// ExternalAttention_68324339744785
// MI455X (gfx1250) — compile-verified, matrix operand order checked by reading, not run on hardware
//
#include <hip/hip_runtime.h>
#include <hip/hip_bf16.h>
#include <math.h>

typedef __attribute__((ext_vector_type(16))) _Float16 v16h;
typedef __attribute__((ext_vector_type(8)))  _Float16 v8h;
typedef __attribute__((ext_vector_type(2)))  _Float16 v2h;
typedef __attribute__((ext_vector_type(8)))  float    v8f;
typedef __attribute__((ext_vector_type(2)))  float    f32x2;

#define BATCH   8
#define CIN     256
#define CI      144   // C_inter
#define CIP     160   // C_inter padded to 5*32 for WMMA K-loop
#define COUT    256
#define HW      16384
#define NHEADS  8
#define CH      18    // channels per head

union F16Frag { v16h v; v8h h[2]; };

static __device__ __forceinline__ v8h ld_v8h(const _Float16* p) {
    return *(const v8h*)p;
}

// ---------------------------------------------------------------------------
// K0: fold BN into k (f16), bias vector, pad v to f16 [COUT x CIP]
// ---------------------------------------------------------------------------
__global__ __launch_bounds__(256) void prep_kernel(
    const float* __restrict__ gamma, const float* __restrict__ beta,
    const float* __restrict__ mean,  const float* __restrict__ var,
    const float* __restrict__ k,     const float* __restrict__ v,
    _Float16* __restrict__ kf, _Float16* __restrict__ v16,
    float* __restrict__ bias)
{
    __shared__ float s_scale[CIN], s_shift[CIN];
    const int t = threadIdx.x;
    if (t < CIN) {
        float sc = gamma[t] * rsqrtf(var[t] + 1e-5f);
        s_scale[t] = sc;
        s_shift[t] = beta[t] - mean[t] * sc;
    }
    __syncthreads();
    for (int idx = t; idx < CI * CIN; idx += 256)
        kf[idx] = (_Float16)(k[idx] * s_scale[idx & (CIN - 1)]);
    for (int idx = t; idx < COUT * CIP; idx += 256) {
        int i = idx % CIP;
        v16[idx] = (i < CI) ? (_Float16)v[(idx / CIP) * CI + i] : (_Float16)0.0f;
    }
    if (t < CI) {
        float s = 0.f;
        for (int c = 0; c < CIN; ++c) s += k[t * CIN + c] * s_shift[c];
        bias[t] = s;
    }
}

// ---------------------------------------------------------------------------
// K1: y[b][144][HW] = kf(144x256,f16) @ f16(xn) + bias      (WMMA f16)
// Block: 256 spatial cols, 8 waves; each wave owns 32 cols (2 B-frags) x all
// 9 M-tiles -> A-fragment traffic halved vs 16-col waves.
// K staged in four 64-channel quarters; LDS tile [s][c] f16, pitch 72 halfs
// (144B) => conflict-free ds_load_b128 B-fragment loads.
// x loads are non-temporal (pure stream; keep y/rbuf resident in L2).
// ---------------------------------------------------------------------------
#define SPITCH 72
__global__ __launch_bounds__(256) void gemm1_kernel(
    const float* __restrict__ x, const _Float16* __restrict__ kf,
    const float* __restrict__ bias, float* __restrict__ y)
{
    __shared__ _Float16 xs[256 * SPITCH];   // ~36.9 KB
    __shared__ float s_bias[CI];
    const int t = threadIdx.x;
    const int wave = t >> 5, lane = t & 31;
    const int b = blockIdx.y;
    const int s0 = blockIdx.x * 256;
    const float* xb = x + ((size_t)b * CIN) * HW;

    if (t < CI) s_bias[t] = bias[t];

    const int nloc0 = (wave << 5) + (lane & 15);  // LDS spatial row, frag 0
    const int nloc1 = nloc0 + 16;                 // frag 1
    const int khb   = (lane < 16) ? 0 : 8;        // K-half base per ISA layout
    v8f acc[9][2] = {};

    for (int kh = 0; kh < 4; ++kh) {              // four K quarters of 64 ch
        const int cbase = kh * 64;
        __syncthreads();  // protect LDS reuse (also covers s_bias on pass 0)
        // stage + transpose + convert: 64 channels x 256 spatial (NT loads)
        for (int idx = t; idx < 64 * 128; idx += 256) {
            int cl = idx >> 7;             // local channel 0..63
            int sp = (idx & 127) << 1;     // spatial pair 0..254
            f32x2 xv = __builtin_nontemporal_load(
                (const f32x2*)(xb + (size_t)(cbase + cl) * HW + s0 + sp));
            xs[(sp    ) * SPITCH + cl] = (_Float16)xv.x;
            xs[(sp + 1) * SPITCH + cl] = (_Float16)xv.y;
        }
        __syncthreads();
        #pragma unroll
        for (int kt = 0; kt < 2; ++kt) {
            const int kb = kt * 32;
            F16Frag bf0, bf1;
            bf0.h[0] = ld_v8h(&xs[nloc0 * SPITCH + kb + khb]);
            bf0.h[1] = ld_v8h(&xs[nloc0 * SPITCH + kb + khb + 16]);
            bf1.h[0] = ld_v8h(&xs[nloc1 * SPITCH + kb + khb]);
            bf1.h[1] = ld_v8h(&xs[nloc1 * SPITCH + kb + khb + 16]);
            const _Float16* ka = kf + (size_t)(lane & 15) * CIN + cbase + kb + khb;
            #pragma unroll
            for (int mt = 0; mt < 9; ++mt) {
                F16Frag af;
                af.h[0] = ld_v8h(ka + (size_t)mt * 16 * CIN);
                af.h[1] = ld_v8h(ka + (size_t)mt * 16 * CIN + 16);
                acc[mt][0] = __builtin_amdgcn_wmma_f32_16x16x32_f16(
                    false, af.v, false, bf0.v, (short)0, acc[mt][0], false, false);
                acc[mt][1] = __builtin_amdgcn_wmma_f32_16x16x32_f16(
                    false, af.v, false, bf1.v, (short)0, acc[mt][1], false, false);
            }
        }
    }
    // epilogue: D layout -> VGPR r: lanes<16 M=r, lanes>=16 M=8+r
    const int mofs = (lane < 16) ? 0 : 8;
    #pragma unroll
    for (int nidx = 0; nidx < 2; ++nidx) {
        const int s = s0 + nloc0 + nidx * 16;
        #pragma unroll
        for (int mt = 0; mt < 9; ++mt) {
            #pragma unroll
            for (int r = 0; r < 8; ++r) {
                int M = mt * 16 + mofs + r;
                y[(((size_t)b * CI + M) << 14) + s] = acc[mt][nidx][r] + s_bias[M];
            }
        }
    }
}

// ---------------------------------------------------------------------------
// K2: per-(b,i) row: spatial max and 1/sum(exp(y-max))  -> stats[row][2]
// (y is L2-resident; regular temporal loads)
// ---------------------------------------------------------------------------
__global__ __launch_bounds__(256) void stats_kernel(
    const float* __restrict__ y, float* __restrict__ stats)
{
    __shared__ float red[256];
    const int row = blockIdx.x;                 // b*CI + i
    const float* yr = y + ((size_t)row << 14);
    const int t = threadIdx.x;
    float m = -3.4e38f;
    for (int s = t; s < HW; s += 256) m = fmaxf(m, yr[s]);
    red[t] = m; __syncthreads();
    for (int off = 128; off > 0; off >>= 1) {
        if (t < off) red[t] = fmaxf(red[t], red[t + off]);
        __syncthreads();
    }
    m = red[0]; __syncthreads();
    float sum = 0.f;
    for (int s = t; s < HW; s += 256) sum += __expf(yr[s] - m);
    red[t] = sum; __syncthreads();
    for (int off = 128; off > 0; off >>= 1) {
        if (t < off) red[t] += red[t + off];
        __syncthreads();
    }
    if (t == 0) { stats[row * 2] = m; stats[row * 2 + 1] = 1.f / red[0]; }
}

// ---------------------------------------------------------------------------
// K3: q = exp(y-max)/Z;  D[h,s] = sum_ch q;  write q/(D+1e-6) as f16 into
// rbuf[b][s][CIP] (K-contiguous, zero padded 144..159) -> GEMM2 B layout.
// ---------------------------------------------------------------------------
#define QPITCH 152
__global__ __launch_bounds__(256) void actdn_kernel(
    const float* __restrict__ y, const float* __restrict__ stats,
    _Float16* __restrict__ rbuf)
{
    __shared__ _Float16 qs[128 * QPITCH];   // ~38.9 KB
    __shared__ float dinv[128][NHEADS];
    const int t  = threadIdx.x;
    const int b  = blockIdx.y;
    const int s0 = blockIdx.x * 128;
    const int sl = t & 127;
    const int hh = (t >> 7) * 4;            // this thread handles 4 heads
    const int s  = s0 + sl;
    const float* yb = y + (((size_t)b * CI) << 14);
    const float* st = stats + (size_t)b * CI * 2;
    for (int h = hh; h < hh + 4; ++h) {
        float d = 0.f;
        for (int j = 0; j < CH; ++j) {
            int i = h * CH + j;
            float q = __expf(yb[((size_t)i << 14) + s] - st[i * 2]) * st[i * 2 + 1];
            qs[sl * QPITCH + i] = (_Float16)q;
            d += q;
        }
        dinv[sl][h] = 1.f / (d + 1e-6f);
    }
    __syncthreads();
    _Float16* rb = rbuf + (((size_t)b << 14) + s0) * CIP;
    for (int idx = t; idx < 128 * (CIP / 2); idx += 256) {
        int srow = idx / (CIP / 2);
        int i0   = (idx % (CIP / 2)) * 2;    // even; pair never crosses a head
        v2h o;
        if (i0 < CI) {
            float dsc = dinv[srow][i0 / CH];
            o.x = (_Float16)((float)qs[srow * QPITCH + i0]     * dsc);
            o.y = (_Float16)((float)qs[srow * QPITCH + i0 + 1] * dsc);
        } else {
            o.x = (_Float16)0.0f; o.y = (_Float16)0.0f;
        }
        *(v2h*)(rb + (size_t)srow * CIP + i0) = o;
    }
}

// ---------------------------------------------------------------------------
// K4: out[b][256][HW] = v16(256xCIP,f16) @ rbuf  (WMMA f16, K=160=5*32)
// Block: 128 cols, 8 waves = 2 O-halves x 4 N-groups; each wave: 8 O-tiles x
// 32 cols (2 B-frags). Fragment traffic per block: 320KB A + 80KB B (vs
// 640+40 for 16-col waves). Output stores are non-temporal (pure stream).
// ---------------------------------------------------------------------------
__global__ __launch_bounds__(256) void gemm2_kernel(
    const _Float16* __restrict__ rbuf, const _Float16* __restrict__ v16,
    float* __restrict__ out)
{
    const int t = threadIdx.x;
    const int wave = t >> 5, lane = t & 31;
    const int b  = blockIdx.y;
    const int s0 = blockIdx.x * 128;
    const int ogrp  = wave & 1;                    // O half: tiles 0..7 / 8..15
    const int nbase = (wave >> 1) * 32;            // 4 groups x 32 columns
    const int sA  = s0 + nbase + (lane & 15);
    const int khb = (lane < 16) ? 0 : 8;
    const _Float16* rrow0 = rbuf + (((size_t)b << 14) + sA) * CIP + khb;
    const _Float16* rrow1 = rrow0 + (size_t)16 * CIP;
    const _Float16* va    = v16 + (size_t)((ogrp * 8) * 16 + (lane & 15)) * CIP + khb;
    v8f acc[8][2] = {};
    #pragma unroll
    for (int kt = 0; kt < 5; ++kt) {
        const int kb = kt * 32;
        F16Frag bf0, bf1;
        bf0.h[0] = ld_v8h(rrow0 + kb);
        bf0.h[1] = ld_v8h(rrow0 + kb + 16);
        bf1.h[0] = ld_v8h(rrow1 + kb);
        bf1.h[1] = ld_v8h(rrow1 + kb + 16);
        #pragma unroll
        for (int ot = 0; ot < 8; ++ot) {
            F16Frag af;
            af.h[0] = ld_v8h(va + (size_t)(ot * 16) * CIP + kb);
            af.h[1] = ld_v8h(va + (size_t)(ot * 16) * CIP + kb + 16);
            acc[ot][0] = __builtin_amdgcn_wmma_f32_16x16x32_f16(
                false, af.v, false, bf0.v, (short)0, acc[ot][0], false, false);
            acc[ot][1] = __builtin_amdgcn_wmma_f32_16x16x32_f16(
                false, af.v, false, bf1.v, (short)0, acc[ot][1], false, false);
        }
    }
    const int mofs = (lane < 16) ? 0 : 8;
    #pragma unroll
    for (int nidx = 0; nidx < 2; ++nidx) {
        float* ob = out + (((size_t)b * COUT) << 14) + sA + nidx * 16;
        #pragma unroll
        for (int ot = 0; ot < 8; ++ot) {
            #pragma unroll
            for (int rr = 0; rr < 8; ++rr) {
                int o = (ogrp * 8 + ot) * 16 + mofs + rr;
                __builtin_nontemporal_store(acc[ot][nidx][rr],
                                            ob + ((size_t)o << 14));
            }
        }
    }
}

// ---------------------------------------------------------------------------
extern "C" void kernel_launch(void* const* d_in, const int* in_sizes, int n_in,
                              void* d_out, int out_size, void* d_ws, size_t ws_size,
                              hipStream_t stream)
{
    (void)in_sizes; (void)n_in; (void)out_size; (void)ws_size;
    const float* x     = (const float*)d_in[0];
    const float* gamma = (const float*)d_in[1];
    const float* beta  = (const float*)d_in[2];
    const float* mean  = (const float*)d_in[3];
    const float* var   = (const float*)d_in[4];
    const float* k     = (const float*)d_in[5];
    const float* v     = (const float*)d_in[6];
    float* out = (float*)d_out;

    char* ws = (char*)d_ws;
    // workspace layout (all offsets 256B aligned), total ~117.6 MB
    float*    yws   = (float*)   (ws);               // 8*144*16384*4 = 75,497,472
    _Float16* rbuf  = (_Float16*)(ws + 75497472);    // 8*16384*160*2 = 41,943,040
    _Float16* kf16  = (_Float16*)(ws + 117440512);   // 144*256*2     =     73,728
    _Float16* v16   = (_Float16*)(ws + 117514240);   // 256*160*2     =     81,920
    float*    bias  = (float*)   (ws + 117596160);   // 144*4
    float*    stats = (float*)   (ws + 117597184);   // 1152*2*4

    prep_kernel <<<1,                256, 0, stream>>>(gamma, beta, mean, var, k, v,
                                                       kf16, v16, bias);
    gemm1_kernel<<<dim3(64,  BATCH), 256, 0, stream>>>(x, kf16, bias, yws);
    stats_kernel<<<BATCH * CI,       256, 0, stream>>>(yws, stats);
    actdn_kernel<<<dim3(128, BATCH), 256, 0, stream>>>(yws, stats, rbuf);
    gemm2_kernel<<<dim3(128, BATCH), 256, 0, stream>>>(rbuf, v16, out);
}
